// AdditiveAttention_11020886081800
// MI455X (gfx1250) — matrix operand order repaired and checked by reading, not compiled, not run on hardware
//
#include <hip/hip_runtime.h>
#include <hip/hip_bf16.h>
#include <math.h>

// ---------------------------------------------------------------------------
// Additive attention, MI455X (gfx1250, wave32, WMMA).
//   B=64, S=2048, H=512
//   energy = tanh(X @ W^T + b)   (never materialized)
//   scores = energy @ v ; softmax over S ; context = weights^T @ X
// Main GEMM runs on v_wmma_f32_16x16x32_bf16 with fp32 accumulation, split
// into two independent accumulation chains for WMMA-pipe ILP. tanh uses the
// gfx1250 hardware transcendental when the builtin exists.
// ---------------------------------------------------------------------------

#define BATCH 64
#define SEQ   2048
#define HID   512
#define ROWS  (BATCH * SEQ)          // 131072
#define TILE_M 64                    // rows of X staged in LDS per block

typedef __attribute__((ext_vector_type(16))) __bf16 v16bf;
typedef __attribute__((ext_vector_type(8)))  __bf16 v8bf;
typedef __attribute__((ext_vector_type(8)))  float  v8f;

__device__ __forceinline__ unsigned short f32_to_bf16_rne(float f) {
    unsigned int u = __float_as_uint(f);
    unsigned int r = u + 0x7FFFu + ((u >> 16) & 1u);   // round-to-nearest-even
    return (unsigned short)(r >> 16);
}

// gfx1250 has a hardware v_tanh_f32; fall back to a 2-transcendental approx.
__device__ __forceinline__ float fast_tanhf(float x) {
#if __has_builtin(__builtin_amdgcn_tanhf)
    return __builtin_amdgcn_tanhf(x);
#elif __has_builtin(__builtin_amdgcn_tanh_f32)
    return __builtin_amdgcn_tanh_f32(x);
#else
    float e = __expf(2.0f * x);                        // v_exp_f32
    return 1.0f - 2.0f / (e + 1.0f);                   // v_rcp_f32 + FMAs
#endif
}

// ------------------------- kernel 1: W fp32 -> bf16 -------------------------
__global__ __launch_bounds__(256)
void convert_w_kernel(const float* __restrict__ W, unsigned short* __restrict__ Wbf) {
    int idx = (blockIdx.x * 256 + threadIdx.x) * 4;      // 262144 elems, 4/thread
    float4 f = *(const float4*)(W + idx);
    unsigned long long p =  (unsigned long long)f32_to_bf16_rne(f.x)
                         | ((unsigned long long)f32_to_bf16_rne(f.y) << 16)
                         | ((unsigned long long)f32_to_bf16_rne(f.z) << 32)
                         | ((unsigned long long)f32_to_bf16_rne(f.w) << 48);
    *(unsigned long long*)(Wbf + idx) = p;
}

// ------------- kernel 2: fused  tanh(X W^T + b) . v  ->  scores -------------
// block = 128 threads (4 waves). Block handles TILE_M=64 rows; wave w owns 16.
__global__ __launch_bounds__(128)
void attn_scores_kernel(const float* __restrict__ X,
                        const unsigned short* __restrict__ Wbf,
                        const float* __restrict__ Wb,
                        const float* __restrict__ Vw,
                        float* __restrict__ scores) {
    __shared__ unsigned short lsA[TILE_M * HID];         // 64 KB, bf16 tile of X

    const int tid  = threadIdx.x;
    const long long row0 = (long long)blockIdx.x * TILE_M;

    // ---- phase 1: global fp32 -> LDS bf16 (coalesced float4 loads) ----
    {
        const float4* Xb = (const float4*)(X + row0 * HID);
        #pragma unroll
        for (int it = 0; it < (TILE_M * HID / 4) / 128; ++it) {
            int idx = tid + it * 128;                    // float4 index
            float4 f = Xb[idx];
            unsigned long long p =  (unsigned long long)f32_to_bf16_rne(f.x)
                                 | ((unsigned long long)f32_to_bf16_rne(f.y) << 16)
                                 | ((unsigned long long)f32_to_bf16_rne(f.z) << 32)
                                 | ((unsigned long long)f32_to_bf16_rne(f.w) << 48);
            *(unsigned long long*)(lsA + idx * 4) = p;
        }
    }
    __syncthreads();

    const int lane  = tid & 31;
    const int wave  = tid >> 5;
    const int laneM = lane & 15;      // A: M row / B,C: N col within tile
    const int laneH = lane >> 4;      // K-half selector
    const int wrow  = wave * 16;

    float sacc[8];
    #pragma unroll
    for (int i = 0; i < 8; ++i) sacc[i] = 0.0f;

    // A fragment base for this lane (CDNA5 16-bit A layout):
    //   lanes 0-15 : M=lane,    K = k*32 + {0..7, 16..23}
    //   lanes 16-31: M=lane-16, K = k*32 + {8..15, 24..31}
    const unsigned short* aRow = lsA + (wrow + laneM) * HID + laneH * 8;

    // Preload all 16 A fragments into registers (compiler keeps ~128 VGPRs).
    v16bf afrag[HID / 32];
    #pragma unroll
    for (int k = 0; k < HID / 32; ++k) {
        const unsigned short* ap = aRow + k * 32;
        v8bf alo = *(const v8bf*)ap;                     // K = k*32 + laneH*8 + 0..7
        v8bf ahi = *(const v8bf*)(ap + 16);              // K = k*32 + laneH*8 + 16..23
        afrag[k] = __builtin_shufflevector(alo, ahi,
                     0,1,2,3,4,5,6,7,8,9,10,11,12,13,14,15);
    }

    for (int ntile = 0; ntile < HID / 16; ++ntile) {     // 32 N-tiles
        const int n = ntile * 16 + laneM;                // output column (CDNA5 B/C layout)
        const unsigned short* bRow = Wbf + (long long)n * HID + laneH * 8;

        // two independent accumulation chains (even/odd K-steps) for ILP
        v8f acc0 = {};
        v8f acc1 = {};
        #pragma unroll
        for (int k = 0; k < HID / 32; k += 2) {          // 16 K-steps, 2 per iter
            const unsigned short* bp0 = bRow + k * 32;
            v8bf b0lo = *(const v8bf*)bp0;
            v8bf b0hi = *(const v8bf*)(bp0 + 16);
            v16bf b0 = __builtin_shufflevector(b0lo, b0hi,
                         0,1,2,3,4,5,6,7,8,9,10,11,12,13,14,15);

            const unsigned short* bp1 = bRow + (k + 1) * 32;
            v8bf b1lo = *(const v8bf*)bp1;
            v8bf b1hi = *(const v8bf*)(bp1 + 16);
            v16bf b1 = __builtin_shufflevector(b1lo, b1hi,
                         0,1,2,3,4,5,6,7,8,9,10,11,12,13,14,15);

            acc0 = __builtin_amdgcn_wmma_f32_16x16x32_bf16(
                       false, afrag[k],     false, b0, (short)0, acc0, false, false);
            acc1 = __builtin_amdgcn_wmma_f32_16x16x32_bf16(
                       false, afrag[k + 1], false, b1, (short)0, acc1, false, false);
        }

        // epilogue: energy[wrow + laneH*8 + i][n] = acc0[i]+acc1[i]
        const float bn = Wb[n];
        const float vn = Vw[n];
        #pragma unroll
        for (int i = 0; i < 8; ++i)
            sacc[i] += fast_tanhf(acc0[i] + acc1[i] + bn) * vn;
    }

    // reduce partial scores across the 16 lanes of each half-wave (column dim)
    #pragma unroll
    for (int off = 1; off < 16; off <<= 1) {
        #pragma unroll
        for (int i = 0; i < 8; ++i)
            sacc[i] += __shfl_xor(sacc[i], off, 32);
    }
    if (laneM == 0) {                                    // lanes 0 and 16 write
        #pragma unroll
        for (int i = 0; i < 8; ++i)
            scores[row0 + wrow + laneH * 8 + i] = sacc[i];
    }
}

// ----------------- kernel 3: masked softmax over S per batch ----------------
__global__ __launch_bounds__(256)
void softmax_kernel(const float* __restrict__ scores,
                    const unsigned char* __restrict__ mask,
                    float* __restrict__ weights) {
    const int b = blockIdx.x;
    const float* srow = scores + (long long)b * SEQ;
    const unsigned char* mrow = mask + (long long)b * SEQ;
    float* wrow = weights + (long long)b * SEQ;

    __shared__ float redMax[8];
    __shared__ float redSum[8];

    float vals[SEQ / 256];
    float mx = -3.4e38f;
    #pragma unroll
    for (int i = 0; i < SEQ / 256; ++i) {
        int s = threadIdx.x + i * 256;
        float v = mrow[s] ? srow[s] : -1e9f;
        vals[i] = v;
        mx = fmaxf(mx, v);
    }
    #pragma unroll
    for (int off = 16; off > 0; off >>= 1)
        mx = fmaxf(mx, __shfl_xor(mx, off, 32));
    if ((threadIdx.x & 31) == 0) redMax[threadIdx.x >> 5] = mx;
    __syncthreads();
    float bmax = redMax[0];
    #pragma unroll
    for (int i = 1; i < 8; ++i) bmax = fmaxf(bmax, redMax[i]);

    float sum = 0.0f;
    #pragma unroll
    for (int i = 0; i < SEQ / 256; ++i) {
        vals[i] = __expf(vals[i] - bmax);
        sum += vals[i];
    }
    #pragma unroll
    for (int off = 16; off > 0; off >>= 1)
        sum += __shfl_xor(sum, off, 32);
    if ((threadIdx.x & 31) == 0) redSum[threadIdx.x >> 5] = sum;
    __syncthreads();
    float bsum = 0.0f;
    #pragma unroll
    for (int i = 0; i < 8; ++i) bsum += redSum[i];
    float inv = 1.0f / bsum;

    #pragma unroll
    for (int i = 0; i < SEQ / 256; ++i)
        wrow[threadIdx.x + i * 256] = vals[i] * inv;
}

// ----------- kernel 4: context partials  ctx[b,h] = sum_s w[b,s] X[b,s,h] ----
#define SCHUNKS 16
__global__ __launch_bounds__(256)
void ctx_partial_kernel(const float* __restrict__ X,
                        const float* __restrict__ weights,
                        float* __restrict__ partial) {
    const int b  = blockIdx.x;                 // 64
    const int sc = blockIdx.y;                 // 16 chunks of 128 seq steps
    const int h2 = threadIdx.x;                // float2 column pair index

    const float* Xb = X + ((long long)b * SEQ + sc * (SEQ / SCHUNKS)) * HID;
    const float* wp = weights + (long long)b * SEQ + sc * (SEQ / SCHUNKS);

    float a0 = 0.0f, a1 = 0.0f;
    for (int s = 0; s < SEQ / SCHUNKS; ++s) {
        float w = wp[s];                       // uniform across block -> scalar load
        float2 x = ((const float2*)(Xb + (long long)s * HID))[h2];
        a0 += w * x.x;
        a1 += w * x.y;
    }
    float* out = partial + ((long long)sc * BATCH + b) * HID;
    out[h2 * 2]     = a0;
    out[h2 * 2 + 1] = a1;
}

__global__ __launch_bounds__(256)
void ctx_reduce_kernel(const float* __restrict__ partial, float* __restrict__ ctx) {
    int i = blockIdx.x * 256 + threadIdx.x;    // 32768 = B*H
    float s = 0.0f;
    #pragma unroll
    for (int c = 0; c < SCHUNKS; ++c)
        s += partial[(long long)c * (BATCH * HID) + i];
    ctx[i] = s;
}

// ---------------------------------------------------------------------------
extern "C" void kernel_launch(void* const* d_in, const int* in_sizes, int n_in,
                              void* d_out, int out_size, void* d_ws, size_t ws_size,
                              hipStream_t stream) {
    const float*         X    = (const float*)d_in[0];          // (B,S,H) fp32
    const unsigned char* mask = (const unsigned char*)d_in[1];  // (B,S) bool
    const float*         W    = (const float*)d_in[2];          // (H,H) fp32
    const float*         Wb   = (const float*)d_in[3];          // (H,)
    const float*         Vw   = (const float*)d_in[4];          // (H,)

    // outputs: [context (B,H) | weights (B,S)]
    float* ctx     = (float*)d_out;
    float* weights = ctx + BATCH * HID;

    // workspace layout: Wbf16 | scores | partials
    unsigned short* Wbf    = (unsigned short*)d_ws;                          // 512 KB
    float*          scores = (float*)((char*)d_ws + HID * HID * 2);          // 512 KB
    float*          part   = scores + ROWS;                                  // 2 MB

    // 1) W -> bf16 (262144 elems, 4/thread)
    convert_w_kernel<<<HID * HID / (256 * 4), 256, 0, stream>>>(W, Wbf);

    // 2) fused WMMA GEMM + tanh + dot(v) -> scores
    attn_scores_kernel<<<ROWS / TILE_M, 128, 0, stream>>>(X, Wbf, Wb, Vw, scores);

    // 3) masked softmax over S -> weights (second output)
    softmax_kernel<<<BATCH, 256, 0, stream>>>(scores, mask, weights);

    // 4) context = weights^T . X  (deterministic partials, no fp atomics)
    ctx_partial_kernel<<<dim3(BATCH, SCHUNKS), 256, 0, stream>>>(X, weights, part);
    ctx_reduce_kernel<<<BATCH * HID / 256, 256, 0, stream>>>(part, ctx);
}
